// SlotAttention_66228395704892
// MI455X (gfx1250) — matrix operand order repaired and checked by reading, not compiled, not run on hardware
//
#include <hip/hip_runtime.h>

typedef _Float16 h1_t;
typedef __attribute__((ext_vector_type(2)))  _Float16 h2_t;
typedef __attribute__((ext_vector_type(8)))  _Float16 v8h;
typedef __attribute__((ext_vector_type(16))) _Float16 v16h;
typedef __attribute__((ext_vector_type(8)))  float    v8f;

__device__ static inline v8f wmma_f16(v16h a, v16h b, v8f c) {
  // D = A(16x32 f16) * B(32x16 f16) + C(16x16 f32)
  return __builtin_amdgcn_wmma_f32_16x16x32_f16(false, a, false, b, (short)0, c,
                                                false, false);
}

__device__ static inline v16h mk16(v8h a0, v8h a1) {
  v16h a;
#pragma unroll
  for (int i = 0; i < 8; ++i) { a[i] = a0[i]; a[i + 8] = a1[i]; }
  return a;
}

// ---------------------------------------------------------------------------
// Pack Wk/Wv [64,768] f32 -> B-fragment f16 layout:
//   idx = ((ch*4 + dt)*32 + lane)*16 + j
//   d = dt*16 + (lane&15);  e = ch*32 + (lane<16?0:16) + j
// ---------------------------------------------------------------------------
__global__ __launch_bounds__(256) void k_pack_w(const float* __restrict__ Wk,
                                                const float* __restrict__ Wv,
                                                h1_t* __restrict__ wkpk,
                                                h1_t* __restrict__ wvpk) {
  int idx = blockIdx.x * 256 + threadIdx.x;
  if (idx >= 24 * 4 * 32 * 16) return;
  int j = idx & 15;
  int lane = (idx >> 4) & 31;
  int dt = (idx >> 9) & 3;
  int ch = idx >> 11;
  int d = dt * 16 + (lane & 15);
  int e = ch * 32 + ((lane < 16) ? 0 : 16) + j;
  wkpk[idx] = (h1_t)Wk[d * 768 + e];
  wvpk[idx] = (h1_t)Wv[d * 768 + e];
}

// ---------------------------------------------------------------------------
// LayerNorm(features) then k = ln(x)*Wk^T, v = ln(x)*Wv^T via WMMA f16.
// 64 rows per WG, 4 waves, ~100KB LDS (320KB/WGP budget).
// ---------------------------------------------------------------------------
__global__ __launch_bounds__(128, 1) void k_ln_kv(
    const float* __restrict__ feats, const float* __restrict__ lng,
    const float* __restrict__ lnb, const h1_t* __restrict__ wkpk,
    const h1_t* __restrict__ wvpk, h1_t* __restrict__ kf16,
    h1_t* __restrict__ vf16) {
  __shared__ h1_t fs[64 * 776];  // padded stride -> conflict-free A reads
  __shared__ h1_t gs[768];
  __shared__ h1_t bs[768];
  __shared__ float mrow[64];
  __shared__ float rrow[64];

  const int t = threadIdx.x;
  const int wave = t >> 5;
  const int lane = t & 31;
  const int b = blockIdx.x >> 6;           // 4096/64 = 64 tiles per batch
  const int n0 = (blockIdx.x & 63) * 64;

  for (int i = t; i < 768; i += 128) {
    gs[i] = (h1_t)lng[i];
    bs[i] = (h1_t)lnb[i];
  }

  {  // per-row stats (2 lanes per row), stage raw f16 into LDS
    const int row = wave * 16 + (lane >> 1);
    const int c0 = (lane & 1) * 384;
    const float* rp = feats + ((size_t)(b * 4096 + n0 + row)) * 768 + c0;
    h1_t* dp = &fs[row * 776 + c0];
    float s = 0.f, ss = 0.f;
#pragma unroll 4
    for (int i = 0; i < 384; i += 4) {
      float4 x = *(const float4*)(rp + i);
      s += x.x + x.y + x.z + x.w;
      ss += x.x * x.x + x.y * x.y + x.z * x.z + x.w * x.w;
      dp[i + 0] = (h1_t)x.x; dp[i + 1] = (h1_t)x.y;
      dp[i + 2] = (h1_t)x.z; dp[i + 3] = (h1_t)x.w;
    }
    s += __shfl_xor(s, 1);
    ss += __shfl_xor(ss, 1);
    if ((lane & 1) == 0) {
      float m = s * (1.0f / 768.0f);
      float var = ss * (1.0f / 768.0f) - m * m;
      mrow[row] = m;
      rrow[row] = rsqrtf(var + 1e-5f);
    }
  }
  __syncthreads();

  const int arow = wave * 16 + (lane & 15);
  const float m = mrow[arow];
  const float r = rrow[arow];
  const h2_t m2 = {(h1_t)m, (h1_t)m};
  const h2_t r2 = {(h1_t)r, (h1_t)r};
  const int koffA = (lane < 16) ? 0 : 16;

  v8f acc[8];  // [0..3] = k tiles, [4..7] = v tiles
  const v8f vz = {0.f, 0.f, 0.f, 0.f, 0.f, 0.f, 0.f, 0.f};
#pragma unroll
  for (int i = 0; i < 8; ++i) acc[i] = vz;

  for (int ch = 0; ch < 24; ++ch) {
    const int e0 = ch * 32;
    const h2_t* pa0 = (const h2_t*)&fs[arow * 776 + e0 + koffA];
    const h2_t* pa1 = (const h2_t*)&fs[arow * 776 + e0 + koffA + 8];
    const h2_t* pg0 = (const h2_t*)&gs[e0 + koffA];
    const h2_t* pg1 = (const h2_t*)&gs[e0 + koffA + 8];
    const h2_t* pb0 = (const h2_t*)&bs[e0 + koffA];
    const h2_t* pb1 = (const h2_t*)&bs[e0 + koffA + 8];
    v16h a;
#pragma unroll
    for (int i = 0; i < 4; ++i) {  // LN affine on packed f16
      h2_t x0 = ((pa0[i] - m2) * r2) * pg0[i] + pb0[i];
      h2_t x1 = ((pa1[i] - m2) * r2) * pg1[i] + pb1[i];
      a[2 * i + 0] = x0[0]; a[2 * i + 1] = x0[1];
      a[2 * i + 8] = x1[0]; a[2 * i + 9] = x1[1];
    }
    const h1_t* wk = wkpk + (size_t)(ch * 4) * 512 + lane * 16;
    const h1_t* wv = wvpk + (size_t)(ch * 4) * 512 + lane * 16;
#pragma unroll
    for (int dt = 0; dt < 4; ++dt) {
      v16h bk = *(const v16h*)(wk + dt * 512);
      v16h bv = *(const v16h*)(wv + dt * 512);
      acc[dt] = wmma_f16(a, bk, acc[dt]);
      acc[4 + dt] = wmma_f16(a, bv, acc[4 + dt]);
    }
  }

  const int mbase = (lane < 16) ? 0 : 8;
  const int col = lane & 15;
#pragma unroll
  for (int dt = 0; dt < 4; ++dt) {
#pragma unroll
    for (int vi = 0; vi < 8; ++vi) {
      const int row = n0 + wave * 16 + mbase + vi;
      const size_t o = ((size_t)(b * 4096 + row)) * 64 + dt * 16 + col;
      kf16[o] = (h1_t)acc[dt][vi];
      vf16[o] = (h1_t)acc[4 + dt][vi];
    }
  }
}

// ---------------------------------------------------------------------------
// Pack v (f16 [B][N][64]) into B-fragment layout [b][chunk][dt][lane][16].
// ---------------------------------------------------------------------------
__global__ __launch_bounds__(256) void k_pack_v(const h1_t* __restrict__ vf16,
                                                h1_t* __restrict__ vpk) {
  int idx = blockIdx.x * 256 + threadIdx.x;  // 32*128*4*32*16 = 8388608
  int j = idx & 15;
  int lane = (idx >> 4) & 31;
  int dt = (idx >> 9) & 3;
  int ch = (idx >> 11) & 127;
  int b = idx >> 18;
  int n = ch * 32 + ((lane < 16) ? 0 : 16) + j;
  int d = dt * 16 + (lane & 15);
  vpk[idx] = vf16[((size_t)(b * 4096 + n)) * 64 + d];
}

__global__ __launch_bounds__(256) void k_init_slots(
    const float* __restrict__ noise, const float* __restrict__ mu,
    const float* __restrict__ sigma, float* __restrict__ slots) {
  int idx = blockIdx.x * 256 + threadIdx.x;
  if (idx >= 32 * 10 * 64) return;
  int d = idx & 63;
  slots[idx] = noise[idx] * sigma[d] + mu[d];
}

// ---------------------------------------------------------------------------
// sn = LN(slots); q = sn * Wq^T, stored f16 with slot rows 10..15 zeroed.
// ---------------------------------------------------------------------------
__global__ __launch_bounds__(64) void k_slotq(const float* __restrict__ slots,
                                              const float* __restrict__ g,
                                              const float* __restrict__ bb,
                                              const float* __restrict__ Wq,
                                              h1_t* __restrict__ q16) {
  const int b = blockIdx.x >> 4, s = blockIdx.x & 15, d = threadIdx.x;
  if (s >= 10) { q16[((size_t)(b * 16 + s)) * 64 + d] = (h1_t)0.f; return; }
  __shared__ float xs[64];
  __shared__ float sn[64];
  float x = slots[((size_t)(b * 10 + s)) * 64 + d];
  xs[d] = x;
  __syncthreads();
  float m = 0.f;
  for (int e = 0; e < 64; ++e) m += xs[e];
  m *= (1.f / 64.f);
  float v = 0.f;
  for (int e = 0; e < 64; ++e) { float dd = xs[e] - m; v += dd * dd; }
  v *= (1.f / 64.f);
  sn[d] = (x - m) * rsqrtf(v + 1e-5f) * g[d] + bb[d];
  __syncthreads();
  float a = 0.f;
  for (int e = 0; e < 64; ++e) a += sn[e] * Wq[d * 64 + e];
  q16[((size_t)(b * 16 + s)) * 64 + d] = (h1_t)a;
}

__global__ __launch_bounds__(256) void k_zero(float* __restrict__ colsum,
                                              float* __restrict__ U) {
  int idx = blockIdx.x * 256 + threadIdx.x;
  if (idx < 512) colsum[idx] = 0.f;
  else if (idx < 33280) U[idx - 512] = 0.f;
}

// ---------------------------------------------------------------------------
// attn = softmax_slots(k*q^T/8) + eps; write attnT f16, accumulate colsum.
// ---------------------------------------------------------------------------
__global__ __launch_bounds__(128, 1) void k_attn(const h1_t* __restrict__ kf16,
                                                 const h1_t* __restrict__ q16,
                                                 h1_t* __restrict__ attnT,
                                                 float* __restrict__ colsum) {
  __shared__ float cs[16];
  const int t = threadIdx.x, wave = t >> 5, lane = t & 31;
  const int b = blockIdx.x >> 5;            // 4096/128 = 32 tiles per batch
  const int n0 = (blockIdx.x & 31) * 128;
  if (t < 16) cs[t] = 0.f;
  __syncthreads();

  const int slot = lane & 15;
  const bool valid = slot < 10;
  const int koffA = (lane < 16) ? 0 : 16;
  const int mbase = (lane < 16) ? 0 : 8;
  const v8f vz = {0.f, 0.f, 0.f, 0.f, 0.f, 0.f, 0.f, 0.f};

  v16h bq[2];
#pragma unroll
  for (int ch = 0; ch < 2; ++ch)
    bq[ch] = *(const v16h*)(q16 + ((size_t)(b * 16 + slot)) * 64 + ch * 32 +
                            ((lane < 16) ? 0 : 16));

  float csl = 0.f;
#pragma unroll
  for (int sub = 0; sub < 2; ++sub) {
    const int nrow = n0 + wave * 32 + sub * 16 + slot;  // A-row for this lane
    v8f acc = vz;
#pragma unroll
    for (int ch = 0; ch < 2; ++ch) {
      const h1_t* pa =
          kf16 + ((size_t)(b * 4096 + nrow)) * 64 + ch * 32 + koffA;
      v16h a = mk16(*(const v8h*)pa, *(const v8h*)(pa + 8));
      acc = wmma_f16(a, bq[ch], acc);
    }
#pragma unroll
    for (int vi = 0; vi < 8; ++vi) {
      float x = acc[vi] * 0.125f;                 // KEY_D^-0.5
      float mx = valid ? x : -1e30f;
#pragma unroll
      for (int msk = 1; msk < 16; msk <<= 1) mx = fmaxf(mx, __shfl_xor(mx, msk));
      float e = valid ? __expf(x - mx) : 0.f;
      float se = e;
#pragma unroll
      for (int msk = 1; msk < 16; msk <<= 1) se += __shfl_xor(se, msk);
      float a = valid ? (e / se + 1e-8f) : 0.f;
      csl += a;
      const int nn = n0 + wave * 32 + sub * 16 + mbase + vi;
      attnT[((size_t)(b * 16 + slot)) * 4096 + nn] = (h1_t)a;
    }
  }
  csl += __shfl_xor(csl, 16);
  if (lane < 16) atomicAdd(&cs[slot], csl);
  __syncthreads();
  if (t < 16) atomicAdd(&colsum[b * 16 + t], cs[t]);
}

// ---------------------------------------------------------------------------
// U = attn^T * v  (token renorm deferred: divide by colsum in k_gru).
// K=4096 split over 8 WGs/batch, 8 waves each.
// ---------------------------------------------------------------------------
__global__ __launch_bounds__(256, 1) void k_updates(
    const h1_t* __restrict__ attnT, const h1_t* __restrict__ vpk,
    float* __restrict__ U) {
  __shared__ float Us[1024];
  const int t = threadIdx.x, wave = t >> 5, lane = t & 31;
  const int b = blockIdx.x >> 3;
  const int split = blockIdx.x & 7;
  for (int i = t; i < 1024; i += 256) Us[i] = 0.f;
  __syncthreads();

  const int slot = lane & 15;
  const int koffA = (lane < 16) ? 0 : 16;
  const v8f vz = {0.f, 0.f, 0.f, 0.f, 0.f, 0.f, 0.f, 0.f};
  v8f acc[4];
#pragma unroll
  for (int i = 0; i < 4; ++i) acc[i] = vz;

#pragma unroll
  for (int i = 0; i < 2; ++i) {
    const int c = split * 16 + wave * 2 + i;  // n-chunk (32 tokens)
    const h1_t* pa =
        attnT + ((size_t)(b * 16 + slot)) * 4096 + c * 32 + koffA;
    v16h a = mk16(*(const v8h*)pa, *(const v8h*)(pa + 8));
    const h1_t* pb = vpk + ((size_t)(b * 128 + c)) * 4 * 512 + lane * 16;
#pragma unroll
    for (int dt = 0; dt < 4; ++dt) {
      v16h bb = *(const v16h*)(pb + dt * 512);
      acc[dt] = wmma_f16(a, bb, acc[dt]);
    }
  }
  const int mbase = (lane < 16) ? 0 : 8;
#pragma unroll
  for (int dt = 0; dt < 4; ++dt)
#pragma unroll
    for (int vi = 0; vi < 8; ++vi)
      atomicAdd(&Us[(mbase + vi) * 64 + dt * 16 + slot], acc[dt][vi]);
  __syncthreads();
  for (int i = t; i < 1024; i += 256) atomicAdd(&U[b * 1024 + i], Us[i]);
}

// ---------------------------------------------------------------------------
// Per (b,slot): renorm updates, GRU (torch gate order r,z,n), LN, MLP+residual.
// ---------------------------------------------------------------------------
__global__ __launch_bounds__(64) void k_gru(
    const float* __restrict__ U, const float* __restrict__ colsum,
    const float* __restrict__ slots_in, const float* __restrict__ W_ih,
    const float* __restrict__ W_hh, const float* __restrict__ b_ih,
    const float* __restrict__ b_hh, const float* __restrict__ g_out,
    const float* __restrict__ b_out, const float* __restrict__ W1,
    const float* __restrict__ b1, const float* __restrict__ W2,
    const float* __restrict__ b2, float* __restrict__ slots_out) {
  const int b = blockIdx.x / 10, s = blockIdx.x % 10, d = threadIdx.x;
  __shared__ float upd[64];
  __shared__ float hprev[64];
  __shared__ float tmp[64];
  __shared__ float h1s[64];

  const float cs = colsum[b * 16 + s];
  upd[d] = U[((size_t)(b * 16 + s)) * 64 + d] / cs;
  const float hp = slots_in[((size_t)(b * 10 + s)) * 64 + d];
  hprev[d] = hp;
  __syncthreads();

  float gi_r = b_ih[d], gi_z = b_ih[64 + d], gi_n = b_ih[128 + d];
  float gh_r = b_hh[d], gh_z = b_hh[64 + d], gh_n = b_hh[128 + d];
  for (int e = 0; e < 64; ++e) {
    const float u = upd[e], h = hprev[e];
    gi_r += u * W_ih[d * 64 + e];
    gi_z += u * W_ih[(64 + d) * 64 + e];
    gi_n += u * W_ih[(128 + d) * 64 + e];
    gh_r += h * W_hh[d * 64 + e];
    gh_z += h * W_hh[(64 + d) * 64 + e];
    gh_n += h * W_hh[(128 + d) * 64 + e];
  }
  const float r = 1.f / (1.f + __expf(-(gi_r + gh_r)));
  const float z = 1.f / (1.f + __expf(-(gi_z + gh_z)));
  const float n = tanhf(gi_n + r * gh_n);
  const float sl = (1.f - z) * n + z * hp;
  tmp[d] = sl;
  __syncthreads();

  float m = 0.f;
  for (int e = 0; e < 64; ++e) m += tmp[e];
  m *= (1.f / 64.f);
  float v = 0.f;
  for (int e = 0; e < 64; ++e) { float dd = tmp[e] - m; v += dd * dd; }
  v *= (1.f / 64.f);
  hprev[d] = (sl - m) * rsqrtf(v + 1e-5f) * g_out[d] + b_out[d];
  __syncthreads();

  float a1 = b1[d];
  for (int e = 0; e < 64; ++e) a1 += hprev[e] * W1[d * 64 + e];
  h1s[d] = fmaxf(a1, 0.f);
  __syncthreads();

  float a2 = b2[d];
  for (int e = 0; e < 64; ++e) a2 += h1s[e] * W2[d * 64 + e];
  slots_out[((size_t)(b * 10 + s)) * 64 + d] = sl + a2;
}

// ---------------------------------------------------------------------------
extern "C" void kernel_launch(void* const* d_in, const int* in_sizes, int n_in,
                              void* d_out, int out_size, void* d_ws,
                              size_t ws_size, hipStream_t stream) {
  const float* feats = (const float*)d_in[0];
  const float* noise = (const float*)d_in[1];
  const float* ln_in_g = (const float*)d_in[2];
  const float* ln_in_b = (const float*)d_in[3];
  const float* ln_sl_g = (const float*)d_in[4];
  const float* ln_sl_b = (const float*)d_in[5];
  const float* ln_o_g = (const float*)d_in[6];
  const float* ln_o_b = (const float*)d_in[7];
  const float* mu = (const float*)d_in[8];
  const float* sigma = (const float*)d_in[9];
  const float* Wk = (const float*)d_in[10];
  const float* Wq = (const float*)d_in[11];
  const float* Wv = (const float*)d_in[12];
  const float* W_ih = (const float*)d_in[13];
  const float* W_hh = (const float*)d_in[14];
  const float* b_ih = (const float*)d_in[15];
  const float* b_hh = (const float*)d_in[16];
  const float* W1 = (const float*)d_in[17];
  const float* b1 = (const float*)d_in[18];
  const float* W2 = (const float*)d_in[19];
  const float* b2 = (const float*)d_in[20];
  (void)in_sizes; (void)n_in; (void)out_size; (void)ws_size;

  unsigned char* ws = (unsigned char*)d_ws;
  size_t off = 0;
  auto wsalloc = [&](size_t bytes) -> void* {
    void* p = ws + off;
    off = (off + bytes + 255) & ~(size_t)255;
    return p;
  };
  h1_t* wkpk = (h1_t*)wsalloc((size_t)49152 * 2);
  h1_t* wvpk = (h1_t*)wsalloc((size_t)49152 * 2);
  h1_t* kf16 = (h1_t*)wsalloc((size_t)32 * 4096 * 64 * 2);
  h1_t* vf16 = (h1_t*)wsalloc((size_t)32 * 4096 * 64 * 2);
  h1_t* vpk  = (h1_t*)wsalloc((size_t)32 * 128 * 4 * 32 * 16 * 2);
  h1_t* q16  = (h1_t*)wsalloc((size_t)32 * 16 * 64 * 2);
  h1_t* attnT = (h1_t*)wsalloc((size_t)32 * 16 * 4096 * 2);
  float* colsum = (float*)wsalloc((size_t)512 * 4);
  float* U = (float*)wsalloc((size_t)32768 * 4);
  float* slots = (float*)wsalloc((size_t)20480 * 4);

  k_pack_w<<<192, 256, 0, stream>>>(Wk, Wv, wkpk, wvpk);
  k_ln_kv<<<2048, 128, 0, stream>>>(feats, ln_in_g, ln_in_b, wkpk, wvpk, kf16,
                                    vf16);
  k_pack_v<<<32768, 256, 0, stream>>>(vf16, vpk);
  k_init_slots<<<80, 256, 0, stream>>>(noise, mu, sigma, slots);

  float* outp = (float*)d_out;
  for (int it = 0; it < 3; ++it) {
    k_slotq<<<512, 64, 0, stream>>>(slots, ln_sl_g, ln_sl_b, Wq, q16);
    k_zero<<<130, 256, 0, stream>>>(colsum, U);
    k_attn<<<1024, 128, 0, stream>>>(kf16, q16, attnT, colsum);
    k_updates<<<256, 256, 0, stream>>>(attnT, vpk, U);
    k_gru<<<320, 64, 0, stream>>>(U, colsum, slots, W_ih, W_hh, b_ih, b_hh,
                                  ln_o_g, ln_o_b, W1, b1, W2, b2,
                                  (it == 2) ? outp : slots);
  }
}
